// RAGModule_74500502716553
// MI455X (gfx1250) — compile-verified, matrix operand order checked by reading, not run on hardware
//
#include <hip/hip_runtime.h>

// ---------------------------------------------------------------------------
// RAG retrieval module for MI455X (gfx1250, wave32, WMMA).
//   1. row-normalize prototypes & queries -> bf16 (protos bf16 = 128MB -> L2)
//   2. fused bf16-WMMA sims GEMM (32 rows/block, 2 P-slices) + top-5
//   3. cross-slice top-5 merge
//   4. softmax(T=10) + gather + aggregate -> combined [B,2D] bf16
//   5. attention gate (WMMA) -> alpha;  fusion MLP (WMMA x2) + residual
// All WMMA loops are software-pipelined one K-chunk ahead so load waits
// overlap matrix math instead of draining to zero before each v_wmma.
// ---------------------------------------------------------------------------

#define B_DIM 4096
#define D_DIM 1024
#define P_DIM 65536
#define NSLICE 2
#define PSLICE (P_DIM / NSLICE)
#define TEMP_F 10.0f

typedef __attribute__((ext_vector_type(4)))  unsigned int v4u;
typedef __attribute__((ext_vector_type(8)))  unsigned int v8u;
typedef __attribute__((ext_vector_type(16))) __bf16      v16bf;
typedef __attribute__((ext_vector_type(8)))  float       v8f;

static __device__ __forceinline__ unsigned short f32_to_bf16(float f) {
  union { float f; unsigned int u; } v; v.f = f;
  unsigned int r = v.u + 0x7FFFu + ((v.u >> 16) & 1u);  // round-nearest-even
  return (unsigned short)(r >> 16);
}

// One 16x16x32 bf16 fragment: 8 bf16 at p, 8 bf16 at p+16 (elements).
// Caller supplies p = row_ptr + k0 + (lane>>4)*8 with row = lane&15.
static __device__ __forceinline__ v16bf load_frag(const unsigned short* p) {
  v4u lo = *(const v4u*)(p);
  v4u hi = *(const v4u*)(p + 16);
  v8u w;
  w[0]=lo[0]; w[1]=lo[1]; w[2]=lo[2]; w[3]=lo[3];
  w[4]=hi[0]; w[5]=hi[1]; w[6]=hi[2]; w[7]=hi[3];
  return __builtin_bit_cast(v16bf, w);
}

static __device__ __forceinline__ v8f wmma_bf16(v16bf a, v16bf b, v8f c) {
  return __builtin_amdgcn_wmma_f32_16x16x32_bf16(
      false, a, false, b, (short)0, c, false, false);
}

// Sorted (descending) 5-element register list insert, all static indices.
static __device__ __forceinline__ void topk_insert(
    float v, int gi,
    float& t0, float& t1, float& t2, float& t3, float& t4,
    int& i0, int& i1, int& i2, int& i3, int& i4) {
  if (v <= t4) return;
  t4 = v; i4 = gi;
  if (t4 > t3) { float f=t3; t3=t4; t4=f; int q=i3; i3=i4; i4=q; }
  if (t3 > t2) { float f=t2; t2=t3; t3=f; int q=i2; i2=i3; i3=q; }
  if (t2 > t1) { float f=t1; t1=t2; t2=f; int q=i1; i1=i2; i2=q; }
  if (t1 > t0) { float f=t0; t0=t1; t1=f; int q=i0; i0=i1; i1=q; }
}

// ---------------------------------------------------------------------------
__global__ void k_cvt_bf16(const float* __restrict__ src,
                           unsigned short* __restrict__ dst, int n) {
  int i = blockIdx.x * blockDim.x + threadIdx.x;
  if (i < n) dst[i] = f32_to_bf16(src[i]);
}

// ---------------------------------------------------------------------------
__global__ __launch_bounds__(256) void k_rownorm_bf16(
    const float* __restrict__ src, unsigned short* __restrict__ dst, int cols) {
  __shared__ float red[8];
  const float* r = src + (size_t)blockIdx.x * cols;
  float s = 0.0f;
  for (int c = threadIdx.x; c < cols; c += 256) { float v = r[c]; s += v * v; }
  for (int off = 16; off > 0; off >>= 1) s += __shfl_down(s, off, 32);
  if ((threadIdx.x & 31) == 0) red[threadIdx.x >> 5] = s;
  __syncthreads();
  if (threadIdx.x == 0) {
    float t = 0.0f;
    for (int i = 0; i < 8; ++i) t += red[i];
    red[0] = t;
  }
  __syncthreads();
  float scale = 1.0f / fmaxf(sqrtf(red[0]), 1e-12f);
  unsigned short* o = dst + (size_t)blockIdx.x * cols;
  for (int c = threadIdx.x; c < cols; c += 256) o[c] = f32_to_bf16(r[c] * scale);
}

// ---------------------------------------------------------------------------
// Fused sims GEMM + top-5.  Block = 32 query rows x one P-slice (32768).
// 8 waves; per 128-col tile each wave computes TWO 16x16 WMMA tiles sharing
// one B fragment (halves L2 B-traffic per FLOP).  grid = (B/32, NSLICE).
// Dynamic LDS layout: qA (32 rows bf16, padded) | tileS (32x132 f32) |
//                     mV (32x40 f32) | mI (32x40 i32)
// ---------------------------------------------------------------------------
#define QA_STRIDE   (D_DIM + 8)
#define TILE_N      128
#define TILE_STRIDE (TILE_N + 4)
#define QA_BYTES    (32 * QA_STRIDE * 2)
#define TILE_BYTES  (32 * TILE_STRIDE * 4)
#define MV_BYTES    (32 * 40 * 4)
#define SIM_LDS_BYTES (QA_BYTES + TILE_BYTES + 2 * MV_BYTES)

__global__ __launch_bounds__(256) void k_sim_topk(
    const unsigned short* __restrict__ protoB,   // [P, D] normalized bf16
    const unsigned short* __restrict__ qB,       // [B, D] normalized bf16
    float* __restrict__ topVs,                   // [NSLICE*B, 5]
    int* __restrict__ topIs) {
  extern __shared__ char smem[];
  unsigned short* qA = (unsigned short*)smem;
  float* tileS = (float*)(smem + QA_BYTES);
  float* mV    = (float*)(smem + QA_BYTES + TILE_BYTES);            // [32][40]
  int*   mI    = (int*)  (smem + QA_BYTES + TILE_BYTES + MV_BYTES); // [32][40]

  const int tid  = threadIdx.x;
  const int lane = tid & 31;
  const int wid  = tid >> 5;
  const int rowBase  = blockIdx.x * 32;
  const int slice    = blockIdx.y;
  const int colStart = slice * PSLICE;

  // Stage 32 query rows into LDS (v4u = 8 bf16 per copy).
  for (int i = tid; i < 32 * (D_DIM / 8); i += 256) {
    int rr = i >> 7;              // D_DIM/8 == 128
    int cc = (i & 127) * 8;
    *(v4u*)(&qA[rr * QA_STRIDE + cc]) =
        *(const v4u*)(qB + (size_t)(rowBase + rr) * D_DIM + cc);
  }
  __syncthreads();

  const int frow = lane & 15;          // fragment row (M for A, N for B)
  const int kh   = (lane >> 4) * 8;    // K-half offset within fragment
  const int half = lane >> 4;

  float t0 = -1e30f, t1 = -1e30f, t2 = -1e30f, t3 = -1e30f, t4 = -1e30f;
  int   i0 = 0, i1 = 0, i2 = 0, i3 = 0, i4 = 0;
  const int krow = tid & 31;   // row this thread tracks top-5 for
  const int ksub = tid >> 5;   // 16-col sub-slice within the 128-col tile

  const unsigned short* aRow0 = qA + frow * QA_STRIDE + kh;
  const unsigned short* aRow1 = aRow0 + 16 * QA_STRIDE;

  for (int t = 0; t < PSLICE / TILE_N; ++t) {
    const int colBase = colStart + t * TILE_N;
    const unsigned short* bRow =
        protoB + (size_t)(colBase + wid * 16 + frow) * D_DIM + kh;

    if (t + 1 < PSLICE / TILE_N)
      __builtin_prefetch((const void*)(bRow + (size_t)TILE_N * D_DIM), 0, 1);

    // Software-pipelined K loop: load chunk k+1 while WMMAs consume chunk k.
    v16bf a0C = load_frag(aRow0);
    v16bf a1C = load_frag(aRow1);
    v16bf bC  = load_frag(bRow);
    v8f acc0 = {}, acc1 = {};
#pragma unroll 8
    for (int k0 = 32; k0 < D_DIM; k0 += 32) {
      v16bf a0N = load_frag(aRow0 + k0);
      v16bf a1N = load_frag(aRow1 + k0);
      v16bf bN  = load_frag(bRow + k0);
      acc0 = wmma_bf16(a0C, bC, acc0);
      acc1 = wmma_bf16(a1C, bC, acc1);
      a0C = a0N; a1C = a1N; bC = bN;
    }
    acc0 = wmma_bf16(a0C, bC, acc0);
    acc1 = wmma_bf16(a1C, bC, acc1);

    // Spill both 16x16 accum tiles into the shared 32x128 sims tile.
    const int cc = wid * 16 + frow;
#pragma unroll
    for (int r = 0; r < 8; ++r) {
      tileS[(r + 8 * half) * TILE_STRIDE + cc]      = acc0[r];
      tileS[(16 + r + 8 * half) * TILE_STRIDE + cc] = acc1[r];
    }
    __syncthreads();

    // Each thread scans 16 columns of its row, updates private top-5.
#pragma unroll
    for (int j = 0; j < 16; ++j) {
      int c = ksub * 16 + j;
      float v = tileS[krow * TILE_STRIDE + c];
      topk_insert(v, colBase + c, t0, t1, t2, t3, t4, i0, i1, i2, i3, i4);
    }
    __syncthreads();
  }

  // Merge 8 partial top-5 lists per row -> slice top-5.
  mV[krow * 40 + ksub * 5 + 0] = t0;  mI[krow * 40 + ksub * 5 + 0] = i0;
  mV[krow * 40 + ksub * 5 + 1] = t1;  mI[krow * 40 + ksub * 5 + 1] = i1;
  mV[krow * 40 + ksub * 5 + 2] = t2;  mI[krow * 40 + ksub * 5 + 2] = i2;
  mV[krow * 40 + ksub * 5 + 3] = t3;  mI[krow * 40 + ksub * 5 + 3] = i3;
  mV[krow * 40 + ksub * 5 + 4] = t4;  mI[krow * 40 + ksub * 5 + 4] = i4;
  __syncthreads();
  if (tid < 32) {
    size_t outRow = (size_t)slice * B_DIM + rowBase + tid;
    for (int j = 0; j < 5; ++j) {
      float best = -1e30f; int bi = 0, bs = 0;
      for (int e = 0; e < 40; ++e) {
        float vv = mV[tid * 40 + e];
        if (vv > best) { best = vv; bi = mI[tid * 40 + e]; bs = e; }
      }
      mV[tid * 40 + bs] = -1e30f;
      topVs[outRow * 5 + j] = best;
      topIs[outRow * 5 + j] = bi;
    }
  }
}

// ---------------------------------------------------------------------------
// Merge the NSLICE sorted top-5 lists per row into the final sorted top-5.
// ---------------------------------------------------------------------------
__global__ void k_topk_merge(const float* __restrict__ tVs,
                             const int* __restrict__ tIs,
                             float* __restrict__ oV, int* __restrict__ oI) {
  int b = blockIdx.x * blockDim.x + threadIdx.x;
  if (b >= B_DIM) return;
  const float* v0 = tVs + (size_t)b * 5;
  const float* v1 = tVs + ((size_t)B_DIM + b) * 5;
  const int*   x0 = tIs + (size_t)b * 5;
  const int*   x1 = tIs + ((size_t)B_DIM + b) * 5;
  int p0 = 0, p1 = 0;
#pragma unroll
  for (int j = 0; j < 5; ++j) {
    float a = (p0 < 5) ? v0[p0] : -1e30f;
    float c = (p1 < 5) ? v1[p1] : -1e30f;
    if (a >= c) { oV[(size_t)b * 5 + j] = a; oI[(size_t)b * 5 + j] = x0[p0]; ++p0; }
    else        { oV[(size_t)b * 5 + j] = c; oI[(size_t)b * 5 + j] = x1[p1]; ++p1; }
  }
}

// ---------------------------------------------------------------------------
// softmax(top5*10) + gather raw fp32 prototypes + aggregate,
// emit combined = [q | agg] as bf16 [B, 2048]. One block per query row.
// ---------------------------------------------------------------------------
__global__ __launch_bounds__(256) void k_aggregate(
    const float* __restrict__ q, const float* __restrict__ protos,
    const float* __restrict__ topV, const int* __restrict__ topI,
    unsigned short* __restrict__ comb) {
  const int b = blockIdx.x;
  float v[5]; int id[5];
#pragma unroll
  for (int k = 0; k < 5; ++k) {
    v[k]  = topV[(size_t)b * 5 + k];
    id[k] = topI[(size_t)b * 5 + k];
  }
  float m = v[0];  // list is sorted descending
  float w[5]; float s = 0.0f;
#pragma unroll
  for (int k = 0; k < 5; ++k) { w[k] = __expf((v[k] - m) * TEMP_F); s += w[k]; }
  float inv = 1.0f / s;
#pragma unroll
  for (int k = 0; k < 5; ++k) w[k] *= inv;

  for (int d = threadIdx.x; d < D_DIM; d += 256) {
    float a = 0.0f;
#pragma unroll
    for (int k = 0; k < 5; ++k) a += w[k] * protos[(size_t)id[k] * D_DIM + d];
    comb[(size_t)b * (2 * D_DIM) + d]         = f32_to_bf16(q[(size_t)b * D_DIM + d]);
    comb[(size_t)b * (2 * D_DIM) + D_DIM + d] = f32_to_bf16(a);
  }
}

// ---------------------------------------------------------------------------
// Attention gate: h = relu(comb @ Wa1^T + ba1), alpha = sigmoid(h.wa2 + ba2).
// Block = 16 rows; 8 waves x one 16-col tile = all 128 cols. Pipelined K loop.
// ---------------------------------------------------------------------------
__global__ __launch_bounds__(256) void k_attn(
    const unsigned short* __restrict__ comb,   // [B, 2048] bf16
    const unsigned short* __restrict__ wa1,    // [128, 2048] bf16
    const float* __restrict__ ba1, const float* __restrict__ wa2,
    const float* __restrict__ ba2, float* __restrict__ alpha) {
  __shared__ float hS[16 * 132];
  const int tid = threadIdx.x, lane = tid & 31, wid = tid >> 5;
  const int rowBase = blockIdx.x * 16;
  const int frow = lane & 15, kh = (lane >> 4) * 8, half = lane >> 4;

  const unsigned short* aR = comb + (size_t)(rowBase + frow) * 2048 + kh;
  const unsigned short* bR = wa1  + (size_t)(wid * 16 + frow) * 2048 + kh;
  v16bf aC = load_frag(aR), bC = load_frag(bR);
  v8f acc = {};
#pragma unroll 8
  for (int k0 = 32; k0 < 2048; k0 += 32) {
    v16bf aN = load_frag(aR + k0);
    v16bf bN = load_frag(bR + k0);
    acc = wmma_bf16(aC, bC, acc);
    aC = aN; bC = bN;
  }
  acc = wmma_bf16(aC, bC, acc);

  const int cc = wid * 16 + frow;
  const float bias = ba1[cc];
#pragma unroll
  for (int r = 0; r < 8; ++r)
    hS[(r + 8 * half) * 132 + cc] = fmaxf(acc[r] + bias, 0.0f);
  __syncthreads();
  if (tid < 16) {
    float s = ba2[0];
    for (int j = 0; j < 128; ++j) s += hS[tid * 132 + j] * wa2[j];
    alpha[rowBase + tid] = 1.0f / (1.0f + __expf(-s));
  }
}

// ---------------------------------------------------------------------------
// Fusion layer 1: f = relu(comb @ Wf1^T + bf1) -> bf16 [B, 1024]. K = 2048.
// ---------------------------------------------------------------------------
__global__ __launch_bounds__(256) void k_fuse1(
    const unsigned short* __restrict__ comb,  // [B, 2048] bf16
    const unsigned short* __restrict__ wf1,   // [1024, 2048] bf16
    const float* __restrict__ bf1, unsigned short* __restrict__ fOut) {
  const int tid = threadIdx.x, lane = tid & 31, wid = tid >> 5;
  const int rowBase = blockIdx.x * 16;
  const int frow = lane & 15, kh = (lane >> 4) * 8, half = lane >> 4;
  const unsigned short* aR = comb + (size_t)(rowBase + frow) * 2048 + kh;

  for (int nt = wid; nt < 64; nt += 8) {
    const int n = nt * 16 + frow;
    const unsigned short* bR = wf1 + (size_t)n * 2048 + kh;
    v16bf aC = load_frag(aR), bC = load_frag(bR);
    v8f acc = {};
#pragma unroll 8
    for (int k0 = 32; k0 < 2048; k0 += 32) {
      v16bf aN = load_frag(aR + k0);
      v16bf bN = load_frag(bR + k0);
      acc = wmma_bf16(aC, bC, acc);
      aC = aN; bC = bN;
    }
    acc = wmma_bf16(aC, bC, acc);
    const float bias = bf1[n];
#pragma unroll
    for (int r = 0; r < 8; ++r) {
      int mm = r + 8 * half;
      fOut[(size_t)(rowBase + mm) * 1024 + n] =
          f32_to_bf16(fmaxf(acc[r] + bias, 0.0f));
    }
  }
}

// ---------------------------------------------------------------------------
// Fusion layer 2 + residual: out = q + alpha * (f @ Wf2^T + bf2). K = 1024.
// ---------------------------------------------------------------------------
__global__ __launch_bounds__(256) void k_fuse2(
    const unsigned short* __restrict__ fIn,   // [B, 1024] bf16
    const unsigned short* __restrict__ wf2,   // [1024, 1024] bf16
    const float* __restrict__ bf2, const float* __restrict__ q,
    const float* __restrict__ alpha, float* __restrict__ out) {
  const int tid = threadIdx.x, lane = tid & 31, wid = tid >> 5;
  const int rowBase = blockIdx.x * 16;
  const int frow = lane & 15, kh = (lane >> 4) * 8, half = lane >> 4;
  const unsigned short* aR = fIn + (size_t)(rowBase + frow) * 1024 + kh;

  for (int nt = wid; nt < 64; nt += 8) {
    const int n = nt * 16 + frow;
    const unsigned short* bR = wf2 + (size_t)n * 1024 + kh;
    v16bf aC = load_frag(aR), bC = load_frag(bR);
    v8f acc = {};
#pragma unroll 8
    for (int k0 = 32; k0 < 1024; k0 += 32) {
      v16bf aN = load_frag(aR + k0);
      v16bf bN = load_frag(bR + k0);
      acc = wmma_bf16(aC, bC, acc);
      aC = aN; bC = bN;
    }
    acc = wmma_bf16(aC, bC, acc);
    const float bias = bf2[n];
#pragma unroll
    for (int r = 0; r < 8; ++r) {
      int mm = rowBase + r + 8 * half;
      size_t o = (size_t)mm * 1024 + n;
      out[o] = q[o] + alpha[mm] * (acc[r] + bias);
    }
  }
}

// ---------------------------------------------------------------------------
// Launch
// ---------------------------------------------------------------------------
extern "C" void kernel_launch(void* const* d_in, const int* in_sizes, int n_in,
                              void* d_out, int out_size, void* d_ws, size_t ws_size,
                              hipStream_t stream) {
  const float* q      = (const float*)d_in[0];
  const float* protos = (const float*)d_in[1];
  const float* Wf1    = (const float*)d_in[2];
  const float* bf1    = (const float*)d_in[3];
  const float* Wf2    = (const float*)d_in[4];
  const float* bf2    = (const float*)d_in[5];
  const float* Wa1    = (const float*)d_in[6];
  const float* ba1    = (const float*)d_in[7];
  const float* Wa2    = (const float*)d_in[8];
  const float* ba2    = (const float*)d_in[9];
  float* out = (float*)d_out;

  char* ws = (char*)d_ws;
  size_t off = 0;
  auto take = [&](size_t bytes) -> void* {
    void* p = ws + off;
    off = (off + bytes + 255) & ~(size_t)255;
    return p;
  };
  unsigned short* protoB = (unsigned short*)take((size_t)P_DIM * D_DIM * 2); // 128 MB
  unsigned short* qB     = (unsigned short*)take((size_t)B_DIM * D_DIM * 2); // 8 MB
  unsigned short* wf1B   = (unsigned short*)take((size_t)1024 * 2048 * 2);   // 4 MB
  unsigned short* wf2B   = (unsigned short*)take((size_t)1024 * 1024 * 2);   // 2 MB
  unsigned short* wa1B   = (unsigned short*)take((size_t)128 * 2048 * 2);    // 0.5 MB
  unsigned short* comb   = (unsigned short*)take((size_t)B_DIM * 2048 * 2);  // 16 MB
  unsigned short* fMid   = (unsigned short*)take((size_t)B_DIM * 1024 * 2);  // 8 MB
  float* topVs           = (float*)take((size_t)NSLICE * B_DIM * 5 * 4);
  int*   topIs           = (int*)take((size_t)NSLICE * B_DIM * 5 * 4);
  float* topV            = (float*)take((size_t)B_DIM * 5 * 4);
  int*   topI            = (int*)take((size_t)B_DIM * 5 * 4);
  float* alphaA          = (float*)take((size_t)B_DIM * 4);

  // Phase 1: normalization + precision conversion
  k_rownorm_bf16<<<P_DIM, 256, 0, stream>>>(protos, protoB, D_DIM);
  k_rownorm_bf16<<<B_DIM, 256, 0, stream>>>(q, qB, D_DIM);
  k_cvt_bf16<<<(1024 * 2048 + 255) / 256, 256, 0, stream>>>(Wf1, wf1B, 1024 * 2048);
  k_cvt_bf16<<<(1024 * 1024 + 255) / 256, 256, 0, stream>>>(Wf2, wf2B, 1024 * 1024);
  k_cvt_bf16<<<(128 * 2048 + 255) / 256, 256, 0, stream>>>(Wa1, wa1B, 128 * 2048);

  // Phase 2: fused sims GEMM + top-5 per slice (256 workgroups x 8 waves)
  dim3 gridSim(B_DIM / 32, NSLICE);
  k_sim_topk<<<gridSim, 256, SIM_LDS_BYTES, stream>>>(protoB, qB, topVs, topIs);
  k_topk_merge<<<(B_DIM + 255) / 256, 256, 0, stream>>>(topVs, topIs, topV, topI);

  // Phase 3: softmax + gather + aggregate -> combined bf16
  k_aggregate<<<B_DIM, 256, 0, stream>>>(q, protos, topV, topI, comb);

  // Phase 4: attention gate
  k_attn<<<B_DIM / 16, 256, 0, stream>>>(comb, wa1B, ba1, Wa2, ba2, alphaA);

  // Phase 5: fusion MLP + residual
  k_fuse1<<<B_DIM / 16, 256, 0, stream>>>(comb, wf1B, bf1, fMid);
  k_fuse2<<<B_DIM / 16, 256, 0, stream>>>(fMid, wf2B, bf2, q, alphaA, out);
}
